// MoeGroupMLP_3375844295161
// MI455X (gfx1250) — compile-verified, matrix operand order checked by reading, not run on hardware
//
#include <hip/hip_runtime.h>
#include <hip/hip_bf16.h>

// ---------------- problem constants ----------------
#define T_TOK 4096
#define H_DIM 2048
#define I_DIM 1408
#define E_EXP 8
#define K_TOP 2
#define R_CAP 9216           // sum of per-expert segments padded to 128: <= 8192 + 8*128

typedef __bf16 v16bf __attribute__((ext_vector_type(16)));
typedef __bf16 v8bf  __attribute__((ext_vector_type(8)));
typedef __bf16 v4bf  __attribute__((ext_vector_type(4)));
typedef float  v8f   __attribute__((ext_vector_type(8)));
typedef float  v4f   __attribute__((ext_vector_type(4)));

__device__ __forceinline__ v8f wmma_bf16(v16bf a, v16bf b, v8f c) {
  return __builtin_amdgcn_wmma_f32_16x16x32_bf16(false, a, false, b, (short)0, c,
                                                 false, false);
}

// ---------------------------------------------------------------------------
// K0: deterministic counting sort of 8192 (token, k) rows by expert.
// Segments padded to multiples of 128 rows. One block, 256 threads,
// 32 entries per thread (chunked histogram -> prefix -> stable scatter).
// ---------------------------------------------------------------------------
__global__ __launch_bounds__(256) void k_build(const int* __restrict__ sel,
                                               int* __restrict__ seg,
                                               int* __restrict__ row_token,
                                               int* __restrict__ pos_map) {
  __shared__ int hist[E_EXP][256];
  __shared__ int base[E_EXP][256];
  __shared__ int segs[E_EXP + 1];
  const int tid = threadIdx.x;

  int lh[E_EXP];
#pragma unroll
  for (int e = 0; e < E_EXP; ++e) lh[e] = 0;
#pragma unroll 4
  for (int i = 0; i < 32; ++i) {
    int r = tid * 32 + i;              // topk-major flat index
    int t = r & (T_TOK - 1);
    int k = r >> 12;
    int e = sel[t * K_TOP + k];
    lh[e]++;
  }
#pragma unroll
  for (int e = 0; e < E_EXP; ++e) hist[e][tid] = lh[e];
  __syncthreads();

  if (tid == 0) {
    int off = 0;
    for (int e = 0; e < E_EXP; ++e) {
      segs[e] = off;
      int c = 0;
      for (int j = 0; j < 256; ++j) c += hist[e][j];
      off += (c + 127) & ~127;         // pad segment to multiple of 128
    }
    segs[E_EXP] = off;
  }
  __syncthreads();

  if (tid < E_EXP) {                    // per-expert chunk prefix
    int b = segs[tid];
    for (int j = 0; j < 256; ++j) { base[tid][j] = b; b += hist[tid][j]; }
  }
  // init all rows to "padding"
  for (int i = tid; i < R_CAP; i += 256) row_token[i] = -1;
  __syncthreads();

  int b[E_EXP];
#pragma unroll
  for (int e = 0; e < E_EXP; ++e) b[e] = base[e][tid];
#pragma unroll 4
  for (int i = 0; i < 32; ++i) {
    int r = tid * 32 + i;
    int t = r & (T_TOK - 1);
    int k = r >> 12;
    int e = sel[t * K_TOP + k];
    int p = b[e]++;
    row_token[p] = t;
    pos_map[r] = p;
  }
  if (tid <= E_EXP) seg[tid] = segs[tid];
}

// ---------------------------------------------------------------------------
// K1: gather fp32 hidden rows into grouped bf16 matrix Xg[R_CAP][H].
// ---------------------------------------------------------------------------
__global__ __launch_bounds__(256) void k_gather(const float* __restrict__ hs,
                                                const int* __restrict__ row_token,
                                                __bf16* __restrict__ xg) {
  const int r = blockIdx.x;
  const int tok = row_token[r];
  const int c0 = threadIdx.x * 8;
  v8bf v;
  if (tok >= 0) {
    const v4f* p = (const v4f*)(hs + (size_t)tok * H_DIM + c0);
    v4f a = p[0], b = p[1];
#pragma unroll
    for (int j = 0; j < 4; ++j) { v[j] = (__bf16)a[j]; v[4 + j] = (__bf16)b[j]; }
  } else {
#pragma unroll
    for (int j = 0; j < 8; ++j) v[j] = (__bf16)0.0f;
  }
  *(v8bf*)&xg[(size_t)r * H_DIM + c0] = v;
}

// ---------------------------------------------------------------------------
// Fragment helpers: ISA 16-bit A/B layout (16x32 / 32x16):
// lanes 0-15 hold K {0..7,16..23}, lanes 16-31 hold K {8..15,24..31};
// each half is one 16-byte LDS chunk when tiles are K-contiguous per row.
// ---------------------------------------------------------------------------
#define LDS_STRIDE 40   // bf16 elems; 80 B row stride (16B-aligned, conflict-friendly)

__device__ __forceinline__ v16bf load_frag(const __bf16* row_base, int half) {
  v16bf f;
  v8bf* fp = (v8bf*)&f;
  fp[0] = *(const v8bf*)(row_base + half * 8);
  fp[1] = *(const v8bf*)(row_base + half * 8 + 16);
  return f;
}

// ---------------------------------------------------------------------------
// K3: fused gate/up grouped GEMM.  C[128x128] per block over K=H.
// Double-buffered LDS with register staging: global loads for step s+1 are
// issued before the 16 WMMAs of step s; regs are committed to the other
// buffer afterwards, so s_wait_loadcnt lands behind the WMMA issue.
// B tiles: fp32 -> bf16 convert + transpose, packed 8-byte ds_store_b64.
// ---------------------------------------------------------------------------
__global__ __launch_bounds__(256) void k_gateup(const __bf16* __restrict__ xg,
                                                const float* __restrict__ gate_w,
                                                const float* __restrict__ up_w,
                                                const int* __restrict__ seg,
                                                __bf16* __restrict__ act) {
  __shared__ __align__(16) __bf16 As[2][128][LDS_STRIDE];
  __shared__ __align__(16) __bf16 Bg[2][128][LDS_STRIDE];
  __shared__ __align__(16) __bf16 Bu[2][128][LDS_STRIDE];

  const int m0 = blockIdx.x * 128;
  const int total = seg[E_EXP];
  if (m0 >= total) return;                 // uniform exit, EXEC stays full
  int e = 0;
#pragma unroll
  for (int i = 0; i < E_EXP; ++i)
    if (m0 >= seg[i] && m0 < seg[i + 1]) e = i;

  const int n0 = blockIdx.y * 128;
  const float* wg = gate_w + (size_t)e * H_DIM * I_DIM;
  const float* wu = up_w   + (size_t)e * H_DIM * I_DIM;

  const int tid  = threadIdx.x;
  const int wave = tid >> 5, lane = tid & 31;
  const int half = lane >> 4, lr = lane & 15;

  // staging thread mapping
  const int ar = tid >> 2, as = (tid & 3) * 8;       // A: two 16B chunks
  const int kq = tid & 7, ns = (tid >> 3) * 4;       // B: 4x4 (k x n) micro-tile

  v8bf a_st[2];
  v4f  g_st[4], u_st[4];

  auto stage_load = [&](int kb) {
    a_st[0] = *(const v8bf*)&xg[(size_t)(m0 + ar) * H_DIM + kb + as];
    a_st[1] = *(const v8bf*)&xg[(size_t)(m0 + ar + 64) * H_DIM + kb + as];
#pragma unroll
    for (int kk = 0; kk < 4; ++kk) {
      size_t ro = (size_t)(kb + kq * 4 + kk) * I_DIM + n0 + ns;
      g_st[kk] = *(const v4f*)(wg + ro);
      u_st[kk] = *(const v4f*)(wu + ro);
    }
    if (kb + 32 < H_DIM) {               // prefetch weight stream 2 steps ahead
      size_t ro = (size_t)(kb + 32 + kq * 4) * I_DIM + n0 + ns;
      __builtin_prefetch(wg + ro, 0, 1);
      __builtin_prefetch(wu + ro, 0, 1);
    }
  };
  auto stage_store = [&](int b) {
    *(v8bf*)&As[b][ar][as]      = a_st[0];
    *(v8bf*)&As[b][ar + 64][as] = a_st[1];
#pragma unroll
    for (int n = 0; n < 4; ++n) {
      v4bf tg, tu;
#pragma unroll
      for (int kk = 0; kk < 4; ++kk) { tg[kk] = (__bf16)g_st[kk][n]; tu[kk] = (__bf16)u_st[kk][n]; }
      *(v4bf*)&Bg[b][ns + n][kq * 4] = tg;
      *(v4bf*)&Bu[b][ns + n][kq * 4] = tu;
    }
  };

  v8f accg[8] = {};
  v8f accu[8] = {};

  stage_load(0);
  stage_store(0);
  __syncthreads();

  const int nsteps = H_DIM / 32;
  for (int s = 0; s < nsteps; ++s) {
    const int buf = s & 1;
    const bool more = (s + 1) < nsteps;
    if (more) stage_load((s + 1) * 32);

    v16bf af = load_frag(&As[buf][wave * 16 + lr][0], half);
#pragma unroll
    for (int n = 0; n < 8; ++n) {
      v16bf bg = load_frag(&Bg[buf][n * 16 + lr][0], half);
      accg[n] = wmma_bf16(af, bg, accg[n]);
      v16bf bu = load_frag(&Bu[buf][n * 16 + lr][0], half);
      accu[n] = wmma_bf16(af, bu, accu[n]);
    }

    if (more) stage_store(buf ^ 1);
    __syncthreads();
  }

  // epilogue: silu(gate) * up -> bf16
#pragma unroll
  for (int n = 0; n < 8; ++n) {
    int col = n0 + n * 16 + lr;
#pragma unroll
    for (int j = 0; j < 8; ++j) {
      int row = m0 + wave * 16 + half * 8 + j;
      float g = accg[n][j], u = accu[n][j];
      float si = g / (1.0f + __expf(-g));
      act[(size_t)row * I_DIM + col] = (__bf16)(si * u);
    }
  }
}

// ---------------------------------------------------------------------------
// K4: down projection grouped GEMM over K=I, N=H.  Same pipeline structure.
// ---------------------------------------------------------------------------
__global__ __launch_bounds__(256) void k_down(const __bf16* __restrict__ act,
                                              const float* __restrict__ down_w,
                                              const int* __restrict__ seg,
                                              __bf16* __restrict__ dwn) {
  __shared__ __align__(16) __bf16 As[2][128][LDS_STRIDE];
  __shared__ __align__(16) __bf16 Bs[2][128][LDS_STRIDE];

  const int m0 = blockIdx.x * 128;
  const int total = seg[E_EXP];
  if (m0 >= total) return;
  int e = 0;
#pragma unroll
  for (int i = 0; i < E_EXP; ++i)
    if (m0 >= seg[i] && m0 < seg[i + 1]) e = i;

  const int n0 = blockIdx.y * 128;
  const float* wd = down_w + (size_t)e * I_DIM * H_DIM;

  const int tid  = threadIdx.x;
  const int wave = tid >> 5, lane = tid & 31;
  const int half = lane >> 4, lr = lane & 15;

  const int ar = tid >> 2, as = (tid & 3) * 8;
  const int kq = tid & 7, ns = (tid >> 3) * 4;

  v8bf a_st[2];
  v4f  d_st[4];

  auto stage_load = [&](int kb) {
    a_st[0] = *(const v8bf*)&act[(size_t)(m0 + ar) * I_DIM + kb + as];
    a_st[1] = *(const v8bf*)&act[(size_t)(m0 + ar + 64) * I_DIM + kb + as];
#pragma unroll
    for (int kk = 0; kk < 4; ++kk)
      d_st[kk] = *(const v4f*)(wd + (size_t)(kb + kq * 4 + kk) * H_DIM + n0 + ns);
    if (kb + 32 < I_DIM)
      __builtin_prefetch(wd + (size_t)(kb + 32 + kq * 4) * H_DIM + n0 + ns, 0, 1);
  };
  auto stage_store = [&](int b) {
    *(v8bf*)&As[b][ar][as]      = a_st[0];
    *(v8bf*)&As[b][ar + 64][as] = a_st[1];
#pragma unroll
    for (int n = 0; n < 4; ++n) {
      v4bf td;
#pragma unroll
      for (int kk = 0; kk < 4; ++kk) td[kk] = (__bf16)d_st[kk][n];
      *(v4bf*)&Bs[b][ns + n][kq * 4] = td;
    }
  };

  v8f acc[8] = {};

  stage_load(0);
  stage_store(0);
  __syncthreads();

  const int nsteps = I_DIM / 32;
  for (int s = 0; s < nsteps; ++s) {
    const int buf = s & 1;
    const bool more = (s + 1) < nsteps;
    if (more) stage_load((s + 1) * 32);

    v16bf af = load_frag(&As[buf][wave * 16 + lr][0], half);
#pragma unroll
    for (int n = 0; n < 8; ++n) {
      v16bf bf_ = load_frag(&Bs[buf][n * 16 + lr][0], half);
      acc[n] = wmma_bf16(af, bf_, acc[n]);
    }

    if (more) stage_store(buf ^ 1);
    __syncthreads();
  }

#pragma unroll
  for (int n = 0; n < 8; ++n) {
    int col = n0 + n * 16 + lr;
#pragma unroll
    for (int j = 0; j < 8; ++j) {
      int row = m0 + wave * 16 + half * 8 + j;
      dwn[(size_t)row * H_DIM + col] = (__bf16)acc[n][j];
    }
  }
}

// ---------------------------------------------------------------------------
// K5: unpermute + router-weighted sum -> fp32 output [T][H].
// ---------------------------------------------------------------------------
__global__ __launch_bounds__(256) void k_unperm(const __bf16* __restrict__ dwn,
                                                const float* __restrict__ rw,
                                                const int* __restrict__ pos_map,
                                                float* __restrict__ out) {
  const int t = blockIdx.x;
  const int c0 = threadIdx.x * 8;
  float acc[8] = {};
#pragma unroll
  for (int k = 0; k < K_TOP; ++k) {
    int p = pos_map[k * T_TOK + t];
    float w = rw[t * K_TOP + k];
    v8bf v = *(const v8bf*)&dwn[(size_t)p * H_DIM + c0];
#pragma unroll
    for (int j = 0; j < 8; ++j) acc[j] += w * (float)v[j];
  }
  v4f* o = (v4f*)&out[(size_t)t * H_DIM + c0];
  v4f o0, o1;
#pragma unroll
  for (int j = 0; j < 4; ++j) { o0[j] = acc[j]; o1[j] = acc[4 + j]; }
  o[0] = o0;
  o[1] = o1;
}

// ---------------------------------------------------------------------------
extern "C" void kernel_launch(void* const* d_in, const int* in_sizes, int n_in,
                              void* d_out, int out_size, void* d_ws, size_t ws_size,
                              hipStream_t stream) {
  const float* hs  = (const float*)d_in[0];   // [T,H]
  const float* rw  = (const float*)d_in[1];   // [T,K]
  const int*   sel = (const int*)d_in[2];     // [T,K]
  const float* gw  = (const float*)d_in[3];   // [E,H,I]
  const float* uw  = (const float*)d_in[4];   // [E,H,I]
  const float* dw  = (const float*)d_in[5];   // [E,I,H]
  float* out = (float*)d_out;

  // workspace layout (256B aligned)
  char* ws = (char*)d_ws;
  int*    seg       = (int*)(ws + 0);                                  // 16 ints
  int*    row_token = (int*)(ws + 256);                                // R_CAP ints
  int*    pos_map   = (int*)(ws + 256 + (size_t)R_CAP * 4);            // K*T ints
  size_t  off       = 256 + (size_t)R_CAP * 4 + (size_t)K_TOP * T_TOK * 4;
  __bf16* xg  = (__bf16*)(ws + off);                                   // [R_CAP,H]
  off += (size_t)R_CAP * H_DIM * 2;
  __bf16* act = (__bf16*)(ws + off);                                   // [R_CAP,I]
  off += (size_t)R_CAP * I_DIM * 2;
  __bf16* dwn = (__bf16*)(ws + off);                                   // [R_CAP,H]

  k_build<<<1, 256, 0, stream>>>(sel, seg, row_token, pos_map);
  k_gather<<<R_CAP, 256, 0, stream>>>(hs, row_token, xg);
  k_gateup<<<dim3(R_CAP / 128, I_DIM / 128), 256, 0, stream>>>(xg, gw, uw, seg, act);
  k_down<<<dim3(R_CAP / 128, H_DIM / 128), 256, 0, stream>>>(act, dw, seg, dwn);
  k_unperm<<<T_TOK, 256, 0, stream>>>(dwn, rw, pos_map, out);

  (void)in_sizes; (void)n_in; (void)out_size; (void)ws_size;
}
